// DMSTGCN_53455162966212
// MI455X (gfx1250) — matrix operand order repaired and checked by reading, not compiled, not run on hardware
//
#include <hip/hip_runtime.h>
#include <hip/hip_bf16.h>

typedef __attribute__((ext_vector_type(16))) _Float16     v16h;
typedef __attribute__((ext_vector_type(8)))  float        v8f;
typedef __attribute__((ext_vector_type(8)))  unsigned int v8u;
typedef _Float16 h16;

#define DEV __device__ __forceinline__

constexpr int B_ = 64, N_ = 500, T_ = 12, DAYS_ = 288, DIMS_ = 40;
constexpr int RC_ = 16, DC_ = 16, SC_ = 8, EC_ = 16, OUT_ = 3, LAYERS_ = 4, RF_ = 16;
constexpr int SKIP_CH_ = 304;
constexpr int TMAXC = 240;       // 16 * max Tn (15)
constexpr int KP_ = 512;         // padded K (v) dimension for GEMM operands
constexpr int DP_ = 64;          // padded DIMS (40 -> 64) for P / de operands

// ---------------- WMMA fragment builders (wave32, CDNA5 layouts, 32-bit packed) -------
// 16-bit A 16x32 (ISA 7.12.2): lane L: M=L&15, half=L>>4.  VGPR p holds the K pair
//   p<4 : K = 2p + 8*half       p>=4 : K = 16 + 2(p-4) + 8*half
DEV v16h frag_a_u(const unsigned int* As /*[16][32] k-pairs*/, int kt, int lane) {
  int m = lane & 15, half = lane >> 4;
  v8u f;
#pragma unroll
  for (int p = 0; p < 8; ++p) {
    int kp = kt * 16 + ((p < 4) ? (p + 4 * half) : (8 + (p - 4) + 4 * half));
    f[p] = As[m * 32 + kp];
  }
  return __builtin_bit_cast(v16h, f);
}
// 16-bit B 32x16: lane L: N=L&15, half=L>>4; same K-pair order as A (p<4: 2p+8*half, p>=4: 16+2(p-4)+8*half)
// with a k-major LDS tile (Xt[col][kpair], stride 33 dwords) pairs are contiguous.
DEV v16h frag_b_u(const unsigned int* Xt, int ccbase, int kt, int lane) {
  int nn = lane & 15, half = lane >> 4;
  v8u f;
#pragma unroll
  for (int p = 0; p < 8; ++p)
    f[p] = Xt[(ccbase + nn) * 33 + kt * 16 + ((p < 4) ? (p + 4 * half) : (8 + (p - 4) + 4 * half))];
  return __builtin_bit_cast(v16h, f);
}

// ---------------- start: 1->16 channel conv with left pad to RF ----------------
__global__ void k_start(const float* __restrict__ x0, const float* __restrict__ W,
                        const float* __restrict__ bia, float* __restrict__ xout) {
  int idx = blockIdx.x * blockDim.x + threadIdx.x;
  int total = 4 * B_ * N_ * RF_;
  if (idx >= total) return;
  int t = idx % RF_;
  int n = (idx / RF_) % N_;
  int b = (idx / (RF_ * N_)) % B_;
  int s = idx / (RF_ * N_ * B_);
  float v = (t < RF_ - T_) ? 0.f : x0[(((size_t)b * 4 + s) * N_ + n) * T_ + (t - (RF_ - T_))];
  float* dst = xout + (size_t)s * (B_ * RC_ * N_ * 16);
#pragma unroll
  for (int c = 0; c < RC_; ++c)
    dst[(((size_t)b * RC_ + c) * N_ + n) * 16 + t] = v * W[s * RC_ + c] + bia[s * RC_ + c];
}

// ---------------- adjacency: M[gi,b] = te . core ----------------
__global__ void k_core(const float* __restrict__ emb_t, const int* __restrict__ ind,
                       const float* __restrict__ core, float* __restrict__ Mc) {
  int gi = blockIdx.x / B_, b = blockIdx.x % B_;
  __shared__ float te[DIMS_];
  if (threadIdx.x < DIMS_) te[threadIdx.x] = emb_t[((size_t)gi * DAYS_ + ind[b]) * DIMS_ + threadIdx.x];
  __syncthreads();
  const float* cg = core + (size_t)gi * DIMS_ * DIMS_ * DIMS_;
  float* out = Mc + ((size_t)gi * B_ + b) * DIMS_ * DIMS_;
  for (int jk = threadIdx.x; jk < DIMS_ * DIMS_; jk += blockDim.x) {
    float acc = 0.f;
#pragma unroll
    for (int i = 0; i < DIMS_; ++i) acc += te[i] * cg[i * DIMS_ * DIMS_ + jk];
    out[jk] = acc;
  }
}

// ------- P[gi,b] = se(500x40) x M(40x40), stored zero-padded [512][64] f16 -------
__global__ void k_proj(const float* __restrict__ emb_s, const float* __restrict__ Mc,
                       h16* __restrict__ P) {
  int gi = blockIdx.x / B_, b = blockIdx.x % B_;
  __shared__ float Ms[DIMS_ * DIMS_];
  const float* Min = Mc + ((size_t)gi * B_ + b) * DIMS_ * DIMS_;
  for (int i = threadIdx.x; i < DIMS_ * DIMS_; i += blockDim.x) Ms[i] = Min[i];
  __syncthreads();
  const float* se = emb_s + (size_t)gi * N_ * DIMS_;
  h16* out = P + ((size_t)gi * B_ + b) * (size_t)KP_ * DP_;
  for (int i = threadIdx.x; i < KP_ * DP_; i += blockDim.x) {
    int n = i >> 6, k = i & 63;
    float acc = 0.f;
    if (n < N_ && k < DIMS_) {
#pragma unroll
      for (int j = 0; j < DIMS_; ++j) acc += se[n * DIMS_ + j] * Ms[j * DIMS_ + k];
    }
    out[i] = (h16)acc;
  }
}

// ------- de -> zero-padded [512][64] f16 per gi -------
__global__ void k_pad_de(const float* __restrict__ emb_d, h16* __restrict__ dst) {
  int idx = blockIdx.x * blockDim.x + threadIdx.x;
  int total = 7 * KP_ * DP_;
  if (idx >= total) return;
  int k = idx & 63;
  int col = (idx >> 6) % KP_;
  int gi = idx / (KP_ * DP_);
  float v = (col < N_ && k < DIMS_) ? emb_d[((size_t)gi * N_ + col) * DIMS_ + k] : 0.f;
  dst[idx] = (h16)v;
}

// ------- adp = softmax_row(relu(P x de^T)) via WMMA; block = 16 rows x 512 cols -------
// operands are pre-padded so all staging loads are unconditional dword loads.
__global__ void k_adp(const h16* __restrict__ P, const h16* __restrict__ deh,
                      h16* __restrict__ adp) {
  int rt = blockIdx.x, b = blockIdx.y, gi = blockIdx.z;
  int tid = threadIdx.x, lane = tid & 31, wv = tid >> 5;
  __shared__ unsigned int Pa[16 * 32];        // 16 rows x 64 k (pairs)
  __shared__ unsigned int Dt[4][16 * 33];     // per-wave de tile: 16 cols x 64 k (pairs)
  __shared__ float Z[16 * 512];
  const h16* Pg = P + ((size_t)gi * B_ + b) * (size_t)KP_ * DP_;
  for (int i = tid; i < 512; i += 128) {
    int m = i >> 5, kp = i & 31;
    int row = rt * 16 + m;
    Pa[i] = ((const unsigned int*)(Pg + (size_t)row * DP_))[kp];
  }
  __syncthreads();
  const h16* de = deh + (size_t)gi * (size_t)KP_ * DP_;
  int half = lane >> 4, nn = lane & 15;
  for (int cti = 0; cti < 8; ++cti) {
    int ct = wv * 8 + cti;  // 0..31 column tiles of 16
#pragma unroll
    for (int j = 0; j < 16; ++j) {     // stage 16 cols x 32 kpairs, this wave's region
      int i = lane + 32 * j;
      int cc = i >> 5, kp = i & 31;
      int col = ct * 16 + cc;
      Dt[wv][cc * 33 + kp] = ((const unsigned int*)(de + (size_t)col * DP_))[kp];
    }
    v8f acc = {};
#pragma unroll
    for (int kt = 0; kt < 2; ++kt) {
      v16h a = frag_a_u(Pa, kt, lane);
      v16h bb = frag_b_u(Dt[wv], 0, kt, lane);
      acc = __builtin_amdgcn_wmma_f32_16x16x32_f16(false, a, false, bb, (short)0, acc, false, false);
    }
#pragma unroll
    for (int r = 0; r < 8; ++r) {
      float v = acc[r];
      Z[(r + 8 * half) * 512 + ct * 16 + nn] = v > 0.f ? v : 0.f;
    }
  }
  __syncthreads();
  // row softmax: 8 lanes per row (shfl within wave32)
  int row = tid >> 3, l8 = tid & 7;
  float mx = -1e30f;
  for (int c = l8; c < N_; c += 8) mx = fmaxf(mx, Z[row * 512 + c]);
  for (int o = 1; o < 8; o <<= 1) mx = fmaxf(mx, __shfl_xor(mx, o, 8));
  float sm = 0.f;
  for (int c = l8; c < N_; c += 8) sm += __expf(Z[row * 512 + c] - mx);
  for (int o = 1; o < 8; o <<= 1) sm += __shfl_xor(sm, o, 8);
  float inv = 1.f / sm;
  int grow = rt * 16 + row;
  if (grow < N_) {
    h16* outp = adp + (((size_t)gi * B_ + b) * N_ + grow) * KP_;
    for (int c = l8; c < N_; c += 8) outp[c] = (h16)(__expf(Z[row * 512 + c] - mx) * inv);
    for (int c = N_ + l8; c < KP_; c += 8) outp[c] = (h16)0.f;  // zero pad for GEMM K
  }
}

// -------- zero the v in [500,512) pad rows of a feature buffer (GEMM K padding) -------
__global__ void k_padzero(h16* __restrict__ F, int ncols) {
  int idx = blockIdx.x * blockDim.x + threadIdx.x;
  int total = ncols * (KP_ - N_);
  if (idx >= total) return;
  int col = idx / (KP_ - N_);
  int v = N_ + idx % (KP_ - N_);
  F[(size_t)col * KP_ + v] = (h16)0.f;
}

// -------- gated dilated TCN -> column-major f16 features F[b][col][v], v-stride KP_ ----
__global__ void k_tcn(const float* __restrict__ xin, const float* __restrict__ fW,
                      const float* __restrict__ fB, const float* __restrict__ gW,
                      const float* __restrict__ gB, h16* __restrict__ Fx,
                      int layer, int d, int Tn) {
  int idx = blockIdx.x * blockDim.x + threadIdx.x;
  int total = 4 * B_ * N_ * Tn;
  if (idx >= total) return;
  int n = idx % N_;
  int t = (idx / N_) % Tn;
  int b = (idx / (N_ * Tn)) % B_;
  int s = idx / (N_ * Tn * B_);
  const float* xi = xin + (size_t)s * (B_ * RC_ * N_ * 16) + (size_t)b * RC_ * N_ * 16;
  const float* fw = fW + ((size_t)(s * LAYERS_ + layer) * DC_) * RC_ * 2;
  const float* gw = gW + ((size_t)(s * LAYERS_ + layer) * DC_) * RC_ * 2;
  const float* fb = fB + (s * LAYERS_ + layer) * DC_;
  const float* gb = gB + (s * LAYERS_ + layer) * DC_;
  float xv0[RC_], xv1[RC_];
#pragma unroll
  for (int c = 0; c < RC_; ++c) {
    xv0[c] = xi[((size_t)c * N_ + n) * 16 + t];
    xv1[c] = xi[((size_t)c * N_ + n) * 16 + t + d];
  }
  h16* out = Fx + (size_t)s * ((size_t)B_ * TMAXC * KP_) + (size_t)b * (16 * Tn) * KP_ + n;
#pragma unroll
  for (int dc = 0; dc < DC_; ++dc) {
    float f = fb[dc], g = gb[dc];
#pragma unroll
    for (int c = 0; c < RC_; ++c) {
      f += fw[(dc * RC_ + c) * 2 + 0] * xv0[c] + fw[(dc * RC_ + c) * 2 + 1] * xv1[c];
      g += gw[(dc * RC_ + c) * 2 + 0] * xv0[c] + gw[(dc * RC_ + c) * 2 + 1] * xv1[c];
    }
    out[(size_t)(dc * Tn + t) * KP_] = (h16)(tanhf(f) * (1.f / (1.f + __expf(-g))));
  }
}

// ---------------- skip: conv1x1 16->8 of stream0, transposed into skip buffer ----------
__global__ void k_skip(const h16* __restrict__ Fx0, const float* __restrict__ sW,
                       const float* __restrict__ sB, float* __restrict__ skip,
                       int layer, int Tn, int choff) {
  int idx = blockIdx.x * blockDim.x + threadIdx.x;
  int total = B_ * N_ * Tn;
  if (idx >= total) return;
  int n = idx % N_;
  int t = (idx / N_) % Tn;
  int b = idx / (N_ * Tn);
  const h16* xp = Fx0 + (size_t)b * (16 * Tn) * KP_ + n;
  const float* w = sW + layer * SC_ * DC_;
  const float* bb = sB + layer * SC_;
#pragma unroll
  for (int sc = 0; sc < SC_; ++sc) {
    float acc = bb[sc];
#pragma unroll
    for (int c = 0; c < DC_; ++c) acc += w[sc * DC_ + c] * (float)xp[(size_t)(c * Tn + t) * KP_];
    skip[((size_t)b * SKIP_CH_ + choff + sc * Tn + t) * N_ + n] = acc;
  }
}

// ---- nconv GEMM: Y[col][w] = sum_v A[w][v] * X[col][v]; f16 in, f32 WMMA accumulate ----
// remaining staging guards (row<N_, col<COLS) are wave-uniform -> scalar branches.
__global__ void k_nconv(const h16* __restrict__ A, const h16* __restrict__ X,
                        h16* __restrict__ Y, int COLS) {
  int wt = blockIdx.x;   // row tile (16 w)
  int cb = blockIdx.y;   // col block (64)
  int b = blockIdx.z;
  int tid = threadIdx.x, lane = tid & 31, wv = tid >> 5;
  __shared__ unsigned int As[16 * 32];   // 16 rows x 64 k (pairs)
  __shared__ unsigned int Xt[64 * 33];   // 64 cols x 64 k (pairs), padded stride
  const h16* Ab = A + (size_t)b * N_ * KP_;
  const h16* Xb = X + (size_t)b * (size_t)COLS * KP_;
  int colbase = cb * 64;
  v8f acc = {};
  for (int k0 = 0; k0 < KP_; k0 += 64) {
    for (int i = tid; i < 512; i += 128) {
      int m = i >> 5, kp = i & 31;
      int row = wt * 16 + m;  // uniform across a wave
      As[i] = (row < N_) ? ((const unsigned int*)(Ab + (size_t)row * KP_))[(k0 >> 1) + kp] : 0u;
    }
    for (int i = tid; i < 2048; i += 128) {
      int cc = i >> 5, kp = i & 31;
      int col = colbase + cc;  // uniform across a wave
      Xt[cc * 33 + kp] =
          (col < COLS) ? ((const unsigned int*)(Xb + (size_t)col * KP_))[(k0 >> 1) + kp] : 0u;
    }
    __builtin_prefetch(Ab + (size_t)(wt * 16) * KP_ + k0 + 64, 0, 1);
    __syncthreads();
#pragma unroll
    for (int kt = 0; kt < 2; ++kt) {
      v16h a = frag_a_u(As, kt, lane);
      v16h bb = frag_b_u(Xt, wv * 16, kt, lane);
      acc = __builtin_amdgcn_wmma_f32_16x16x32_f16(false, a, false, bb, (short)0, acc, false, false);
    }
    __syncthreads();
  }
  int half = lane >> 4, nn = lane & 15;
  int col = colbase + wv * 16 + nn;
#pragma unroll
  for (int r = 0; r < 8; ++r) {
    int w = wt * 16 + r + 8 * half;
    if (w < N_ && col < COLS) Y[((size_t)b * COLS + col) * KP_ + w] = (h16)acc[r];
  }
}

// ------- gcn combine: out (+)= W(16x48) . [x;x1;x2] + b ; OUTP layout [b][c][t][n] -----
__global__ void k_combine(const h16* __restrict__ Fx, const h16* __restrict__ F1,
                          const h16* __restrict__ F2, const float* __restrict__ W,
                          const float* __restrict__ bia, float* __restrict__ outp,
                          int Tn, int accflag) {
  __shared__ float Ws[RC_ * 48];
  for (int i = threadIdx.x; i < RC_ * 48; i += blockDim.x) Ws[i] = W[i];
  __syncthreads();
  int idx = blockIdx.x * blockDim.x + threadIdx.x;
  int total = B_ * N_ * Tn;
  if (idx >= total) return;
  int n = idx % N_;
  int t = (idx / N_) % Tn;
  int b = idx / (N_ * Tn);
  size_t fbase = (size_t)b * (16 * Tn) * KP_ + n;
  float acc[RC_];
#pragma unroll
  for (int o = 0; o < RC_; ++o) acc[o] = bia[o];
  for (int cc = 0; cc < 48; ++cc) {
    const h16* src = cc < 16 ? Fx : (cc < 32 ? F1 : F2);
    float v = (float)src[fbase + (size_t)((cc & 15) * Tn + t) * KP_];
#pragma unroll
    for (int o = 0; o < RC_; ++o) acc[o] += Ws[o * 48 + cc] * v;
  }
#pragma unroll
  for (int o = 0; o < RC_; ++o) {
    size_t oi = (((size_t)b * RC_ + o) * Tn + t) * N_ + n;
    if (accflag) outp[oi] += acc[o]; else outp[oi] = acc[o];
  }
}

__global__ void k_zero(float* __restrict__ p, int n) {
  int i = blockIdx.x * blockDim.x + threadIdx.x;
  if (i < n) p[i] = 0.f;
}

// ---------------- residual add + bnorm statistics (OUTP layout [b][c][t][n]) ----------
__global__ void k_resid_stats(float* __restrict__ outp, const float* __restrict__ xcur,
                              float* __restrict__ stats, int Tn, int d) {
  int c = blockIdx.y, s = blockIdx.z;
  float* op = outp + (size_t)s * (B_ * RC_ * N_ * 15);
  const float* xc = xcur + (size_t)s * (B_ * RC_ * N_ * 16);
  int per = N_ * Tn;
  int idx = blockIdx.x * blockDim.x + threadIdx.x;
  float v = 0.f, v2 = 0.f;
  if (idx < B_ * per) {
    int b = idx / per, nt = idx % per;
    int t = nt / N_, n = nt % N_;
    size_t oi = ((size_t)b * RC_ + c) * per + nt;
    float val = op[oi] + xc[(((size_t)b * RC_ + c) * N_ + n) * 16 + d + t];
    op[oi] = val;
    v = val; v2 = val * val;
  }
  __shared__ float s1[256], s2[256];
  s1[threadIdx.x] = v; s2[threadIdx.x] = v2;
  __syncthreads();
  for (int o = 128; o > 0; o >>= 1) {
    if (threadIdx.x < o) { s1[threadIdx.x] += s1[threadIdx.x + o]; s2[threadIdx.x] += s2[threadIdx.x + o]; }
    __syncthreads();
  }
  if (threadIdx.x == 0) {
    atomicAdd(&stats[(s * RC_ + c) * 2 + 0], s1[0]);
    atomicAdd(&stats[(s * RC_ + c) * 2 + 1], s2[0]);
  }
}

__global__ void k_bn(const float* __restrict__ outp, const float* __restrict__ stats,
                     const float* __restrict__ g, const float* __restrict__ bb,
                     float* __restrict__ xnext, int Tn, int layer) {
  int idx = blockIdx.x * blockDim.x + threadIdx.x;
  int total = 4 * B_ * RC_ * N_ * Tn;
  if (idx >= total) return;
  int n = idx % N_;
  int t = (idx / N_) % Tn;
  int c = (idx / (N_ * Tn)) % RC_;
  int b = (idx / (N_ * Tn * RC_)) % B_;
  int s = idx / (N_ * Tn * RC_ * B_);
  float cnt = (float)(B_ * N_ * Tn);
  float m = stats[(s * RC_ + c) * 2 + 0] / cnt;
  float var = stats[(s * RC_ + c) * 2 + 1] / cnt - m * m;
  float sc = rsqrtf(var + 1e-5f) * g[(s * LAYERS_ + layer) * RC_ + c];
  float sh = bb[(s * LAYERS_ + layer) * RC_ + c];
  float v = outp[(size_t)s * (B_ * RC_ * N_ * 15) + (((size_t)b * RC_ + c) * Tn + t) * N_ + n];
  xnext[(size_t)s * (B_ * RC_ * N_ * 16) + (((size_t)b * RC_ + c) * N_ + n) * 16 + t] = (v - m) * sc + sh;
}

// ---------------- head: relu(skip) -> end1 -> relu -> end2 ----------------
__global__ void k_end(const float* __restrict__ skip, const float* __restrict__ W1,
                      const float* __restrict__ b1, const float* __restrict__ W2,
                      const float* __restrict__ b2, float* __restrict__ out) {
  int idx = blockIdx.x * blockDim.x + threadIdx.x;
  if (idx >= B_ * N_) return;
  int n = idx % N_, b = idx / N_;
  float e[EC_];
#pragma unroll
  for (int ec = 0; ec < EC_; ++ec) e[ec] = b1[ec];
  for (int ch = 0; ch < SKIP_CH_; ++ch) {
    float hv = skip[((size_t)b * SKIP_CH_ + ch) * N_ + n];
    hv = hv > 0.f ? hv : 0.f;
#pragma unroll
    for (int ec = 0; ec < EC_; ++ec) e[ec] += W1[ec * SKIP_CH_ + ch] * hv;
  }
#pragma unroll
  for (int o = 0; o < OUT_; ++o) {
    float acc = b2[o];
#pragma unroll
    for (int ec = 0; ec < EC_; ++ec) {
      float r = e[ec] > 0.f ? e[ec] : 0.f;
      acc += W2[o * EC_ + ec] * r;
    }
    out[((size_t)b * OUT_ + o) * N_ + n] = acc;
  }
}

extern "C" void kernel_launch(void* const* d_in, const int* in_sizes, int n_in,
                              void* d_out, int out_size, void* d_ws, size_t ws_size,
                              hipStream_t stream) {
  (void)in_sizes; (void)n_in; (void)out_size; (void)ws_size;
  const float* x0      = (const float*)d_in[0];
  const int*   ind     = (const int*)d_in[1];
  const float* start_W = (const float*)d_in[2];
  const float* start_b = (const float*)d_in[3];
  const float* emb_t   = (const float*)d_in[4];
  const float* emb_s   = (const float*)d_in[5];
  const float* emb_d   = (const float*)d_in[6];
  const float* core    = (const float*)d_in[7];
  const float* filt_W  = (const float*)d_in[8];
  const float* filt_b  = (const float*)d_in[9];
  const float* gate_W  = (const float*)d_in[10];
  const float* gate_b  = (const float*)d_in[11];
  const float* skip_W  = (const float*)d_in[12];
  const float* skip_b  = (const float*)d_in[13];
  const float* gc_W    = (const float*)d_in[14];
  const float* gc_b    = (const float*)d_in[15];
  const float* bn_g    = (const float*)d_in[16];
  const float* bn_b    = (const float*)d_in[17];
  const float* e1W     = (const float*)d_in[18];
  const float* e1b     = (const float*)d_in[19];
  const float* e2W     = (const float*)d_in[20];
  const float* e2b     = (const float*)d_in[21];
  float* out = (float*)d_out;

  char* ws = (char*)d_ws;
  size_t off = 0;
  auto alloc = [&](size_t bytes) -> char* {
    char* p = ws + off;
    off = (off + bytes + 255) & ~(size_t)255;
    return p;
  };
  const size_t FSSZ = (size_t)B_ * TMAXC * KP_;  // halves per feature-stream buffer
  float* XA   = (float*)alloc(4ull * B_ * RC_ * N_ * 16 * 4);
  float* XB   = (float*)alloc(4ull * B_ * RC_ * N_ * 16 * 4);
  h16*   FX   = (h16*)  alloc(4ull * FSSZ * 2);
  h16*   F1   = (h16*)  alloc(FSSZ * 2);
  h16*   F2   = (h16*)  alloc(FSSZ * 2);
  float* OUTP = (float*)alloc(4ull * B_ * RC_ * N_ * 15 * 4);
  h16*   ADP  = (h16*)  alloc(7ull * B_ * N_ * KP_ * 2);
  h16*   PH   = (h16*)  alloc(7ull * B_ * KP_ * DP_ * 2);
  float* MC   = (float*)alloc(7ull * B_ * DIMS_ * DIMS_ * 4);
  h16*   DEH  = (h16*)  alloc(7ull * KP_ * DP_ * 2);
  float* SKIPB= (float*)alloc((size_t)B_ * SKIP_CH_ * N_ * 4);
  float* STATS= (float*)alloc(4 * RC_ * 2 * 4);

  { int tot = 4 * B_ * N_ * RF_;
    k_start<<<(tot + 255) / 256, 256, 0, stream>>>(x0, start_W, start_b, XA); }

  k_core<<<7 * B_, 256, 0, stream>>>(emb_t, ind, core, MC);
  k_proj<<<7 * B_, 256, 0, stream>>>(emb_s, MC, PH);
  { int tot = 7 * KP_ * DP_;
    k_pad_de<<<(tot + 255) / 256, 256, 0, stream>>>(emb_d, DEH); }
  { dim3 g(32, B_, 7);
    k_adp<<<g, 128, 0, stream>>>(PH, DEH, ADP); }

  // zero GEMM v-padding of feature buffers once (never overwritten afterwards)
  { int tot = 4 * B_ * TMAXC * (KP_ - N_);
    k_padzero<<<(tot + 255) / 256, 256, 0, stream>>>(FX, 4 * B_ * TMAXC); }
  { int tot = B_ * TMAXC * (KP_ - N_);
    k_padzero<<<(tot + 255) / 256, 256, 0, stream>>>(F1, B_ * TMAXC);
    k_padzero<<<(tot + 255) / 256, 256, 0, stream>>>(F2, B_ * TMAXC); }

  const int dils[4] = {1, 2, 4, 8};
  const int Tns[4] = {15, 13, 9, 1};
  const int skoffs[4] = {184, 80, 8, 0};
  float* xcur = XA;
  float* xnext = XB;
  for (int i = 0; i < LAYERS_; ++i) {
    int d = dils[i], Tn = Tns[i], COLS = 16 * Tn;
    { int tot = 4 * B_ * N_ * Tn;
      k_tcn<<<(tot + 255) / 256, 256, 0, stream>>>(xcur, filt_W, filt_b, gate_W, gate_b, FX, i, d, Tn); }
    { int tot = B_ * N_ * Tn;
      k_skip<<<(tot + 255) / 256, 256, 0, stream>>>(FX, skip_W, skip_b, SKIPB, i, Tn, skoffs[i]); }
    if (i == LAYERS_ - 1) break;  // last layer's GCN/bnorm outputs are dead
    k_zero<<<1, 128, 0, stream>>>(STATS, 4 * RC_ * 2);
    for (int gi = 0; gi < 7; ++gi) {
      int src = gi < 4 ? gi : gi - 3;
      int dst = gi < 4 ? gi : 0;
      int accf = (gi >= 4) ? 1 : 0;
      const h16* A = ADP + (size_t)gi * B_ * N_ * KP_;
      const h16* Xs = FX + (size_t)src * FSSZ;
      dim3 gg(32, (COLS + 63) / 64, B_);
      k_nconv<<<gg, 128, 0, stream>>>(A, Xs, F1, COLS);
      k_nconv<<<gg, 128, 0, stream>>>(A, F1, F2, COLS);
      const float* W = gc_W + ((size_t)(gi * LAYERS_ + i) * RC_) * 48;
      const float* bb = gc_b + (gi * LAYERS_ + i) * RC_;
      int tot = B_ * N_ * Tn;
      k_combine<<<(tot + 255) / 256, 256, 0, stream>>>(Xs, F1, F2, W, bb,
          OUTP + (size_t)dst * B_ * RC_ * N_ * 15, Tn, accf);
    }
    { dim3 gg((B_ * N_ * Tn + 255) / 256, RC_, 4);
      k_resid_stats<<<gg, 256, 0, stream>>>(OUTP, xcur, STATS, Tn, d); }
    { int tot = 4 * B_ * RC_ * N_ * Tn;
      k_bn<<<(tot + 255) / 256, 256, 0, stream>>>(OUTP, STATS, bn_g, bn_b, xnext, Tn, i); }
    float* tmp = xcur; xcur = xnext; xnext = tmp;
  }
  { int tot = B_ * N_;
    k_end<<<(tot + 255) / 256, 256, 0, stream>>>(SKIPB, e1W, e1b, e2W, e2b, out); }
}
